// MambaLayer_35785667510845
// MI455X (gfx1250) — matrix operand order repaired and checked by reading, not compiled, not run on hardware
//
#include <hip/hip_runtime.h>
#include <cstdint>
#include <cstddef>

// ---------------- problem constants (from reference) ----------------
#define BATCH   2
#define SEQ     2048
#define D_MODEL 1024
#define D_STATE 16
#define D_CONV  4
#define D_INNER 2048
#define DT_RANK 64
#define MROWS   (BATCH * SEQ)        // 4096 "token" rows

typedef __attribute__((ext_vector_type(16))) _Float16 v16h;
typedef __attribute__((ext_vector_type(8)))  float    v8f;
typedef unsigned int u32;
typedef __attribute__((ext_vector_type(4))) u32 u32x4;
typedef __attribute__((ext_vector_type(8))) int  i32x8;
typedef __attribute__((ext_vector_type(4))) int  i32x4;

#if defined(__gfx1250__) && __has_builtin(__builtin_amdgcn_tensor_load_to_lds) && \
    __has_builtin(__builtin_amdgcn_s_wait_tensorcnt)
#define USE_TDM 1
#else
#define USE_TDM 0
#endif

// ---------------------------------------------------------------------
// fragment builders: load f32 as grouped float4 (batched under one wait),
// convert to f16 afterwards.
// A fragment: lane holds M=lr; halves = one contiguous 16-elt K span
//             starting at lhi*16
// B fragment: lane holds N=lr; halves = one contiguous 16-elt K span
//             starting at lhi*16 of row-major W[N][K]
// ---------------------------------------------------------------------
__device__ __forceinline__ v16h frag_a(const float* __restrict__ ar, int lhi)
{
    const float4 f0 = *(const float4*)(ar + lhi * 16);
    const float4 f1 = *(const float4*)(ar + lhi * 16 + 4);
    const float4 f2 = *(const float4*)(ar + lhi * 16 + 8);
    const float4 f3 = *(const float4*)(ar + lhi * 16 + 12);
    v16h a;
    a[0]=(_Float16)f0.x;  a[1]=(_Float16)f0.y;  a[2]=(_Float16)f0.z;  a[3]=(_Float16)f0.w;
    a[4]=(_Float16)f1.x;  a[5]=(_Float16)f1.y;  a[6]=(_Float16)f1.z;  a[7]=(_Float16)f1.w;
    a[8]=(_Float16)f2.x;  a[9]=(_Float16)f2.y;  a[10]=(_Float16)f2.z; a[11]=(_Float16)f2.w;
    a[12]=(_Float16)f3.x; a[13]=(_Float16)f3.y; a[14]=(_Float16)f3.z; a[15]=(_Float16)f3.w;
    return a;
}

__device__ __forceinline__ v16h frag_b(const float* __restrict__ br)
{
    const float4 f0 = *(const float4*)(br);
    const float4 f1 = *(const float4*)(br + 4);
    const float4 f2 = *(const float4*)(br + 8);
    const float4 f3 = *(const float4*)(br + 12);
    v16h b;
    b[0]=(_Float16)f0.x;  b[1]=(_Float16)f0.y;  b[2]=(_Float16)f0.z;  b[3]=(_Float16)f0.w;
    b[4]=(_Float16)f1.x;  b[5]=(_Float16)f1.y;  b[6]=(_Float16)f1.z;  b[7]=(_Float16)f1.w;
    b[8]=(_Float16)f2.x;  b[9]=(_Float16)f2.y;  b[10]=(_Float16)f2.z; b[11]=(_Float16)f2.w;
    b[12]=(_Float16)f3.x; b[13]=(_Float16)f3.y; b[14]=(_Float16)f3.z; b[15]=(_Float16)f3.w;
    return b;
}

#if USE_TDM
// Low 32 bits of the flat shared-aperture address are the LDS byte offset
// (see ISA section 10.2).
__device__ __forceinline__ u32 lds_off(const void* p) { return (u32)(uintptr_t)p; }

// Issue a 2D TDM tile load: rows x 32 f32 elements, row stride ld_elems,
// compacted row-major into LDS at lds_byte_off.  D# layout per
// CDNA5 ISA (group0 + group1; groups 2/3 zero for 2D).
// This toolchain exposes the 6-arg builtin form:
//   (uint32x4 g0, int32x8 g1, int32x4 g2, int32x4 g3, int32x8 gx, i32 cpol)
__device__ __forceinline__ void tdm_issue_2d(const float* __restrict__ gsrc,
                                             u32 ld_elems, u32 rows, u32 lds_byte_off)
{
    const uint64_t ga = (uint64_t)(uintptr_t)gsrc;
    u32x4 g0;
    g0.x = 1u;                                            // count=1 (valid user D#)
    g0.y = lds_byte_off;                                  // lds_addr  [63:32]
    g0.z = (u32)ga;                                       // global_addr[31:0]
    g0.w = (u32)((ga >> 32) & 0x1FFFFFFu) | (2u << 30);   // global_addr[56:32] | type=2
    i32x8 g1;
    g1[0] = (int)(2u << 16);                              // data_size=2 -> 4 bytes
    g1[1] = (int)((ld_elems & 0xFFFFu) << 16);            // tensor_dim0[15:0]
    g1[2] = (int)(((ld_elems >> 16) & 0xFFFFu) |          // tensor_dim0[31:16]
                  ((rows & 0xFFFFu) << 16));              // tensor_dim1[15:0]
    g1[3] = (int)(((rows >> 16) & 0xFFFFu) |              // tensor_dim1[31:16]
                  (32u << 16));                           // tile_dim0 = 32
    g1[4] = (int)(rows & 0xFFFFu);                        // tile_dim1 = rows, tile_dim2=0
    g1[5] = (int)ld_elems;                                // tensor_dim0_stride[31:0]
    g1[6] = 0;                                            // stride hi | dim1_stride lo
    g1[7] = 0;
    const i32x4 gz4 = {0, 0, 0, 0};
    const i32x8 gz8 = {0, 0, 0, 0, 0, 0, 0, 0};
    __builtin_amdgcn_tensor_load_to_lds(g0, g1, gz4, gz4, gz8, 0);
}
#endif

// ======================================================================
// Tiled WMMA GEMM:  C[M x N] = epi( A[M x K] @ W[N x K]^T )
// block = 256 threads = 8 waves; block tile 64(M) x 128(N); K step 32.
// Waves arranged 4(M-tiles) x 2(N-groups); each wave: 16x64 via 4 wmma.
// Tiles staged in LDS; TDM double-buffered when available.
// EPI: 0 = plain, 1 = softplus(c + bias[n])
// ======================================================================
template<int EPI>
__global__ __launch_bounds__(256)
void wmma_gemm_tiled(const float* __restrict__ A, int lda,
                     const float* __restrict__ W, int ldw,
                     float* __restrict__ C, int ldc,
                     const float* __restrict__ bias,
                     int M, int N, int K)
{
    constexpr int BM = 64, BN = 128, BK = 32;
    __shared__ __align__(16) float sA[2][BM * BK];   //  8 KB x2
    __shared__ __align__(16) float sB[2][BN * BK];   // 16 KB x2

    const int tid  = threadIdx.x;
    const int lane = tid & 31;
    const int wv   = tid >> 5;        // 0..7
    const int tml  = wv & 3;          // M tile within block
    const int tgl  = wv >> 2;         // N group (64 wide) within block
    const int lr   = lane & 15;
    const int lhi  = lane >> 4;
    const int bm0  = blockIdx.y * BM;
    const int bn0  = blockIdx.x * BN;

    v8f acc[4];
#pragma unroll
    for (int j = 0; j < 4; ++j) acc[j] = (v8f){0,0,0,0,0,0,0,0};

    const int nsteps = K / BK;

#if USE_TDM
    if (wv == 0) {
        tdm_issue_2d(A + (size_t)bm0 * lda, (u32)lda, BM, lds_off(&sA[0][0]));
        tdm_issue_2d(W + (size_t)bn0 * ldw, (u32)ldw, BN, lds_off(&sB[0][0]));
    }
    for (int s = 0; s < nsteps; ++s) {
        const int cur = s & 1;
        if (wv == 0) __builtin_amdgcn_s_wait_tensorcnt(0);  // tiles for step s landed
        __syncthreads();                                    // publish; also fences step s-1 readers
        if (wv == 0 && (s + 1) < nsteps) {                  // DMA next tiles under compute
            const int k1 = (s + 1) * BK;
            tdm_issue_2d(A + (size_t)bm0 * lda + k1, (u32)lda, BM, lds_off(&sA[cur ^ 1][0]));
            tdm_issue_2d(W + (size_t)bn0 * ldw + k1, (u32)ldw, BN, lds_off(&sB[cur ^ 1][0]));
        }
        const float* Ab = &sA[cur][0];
        const float* Bb = &sB[cur][0];
        const v16h a = frag_a(Ab + (tml * 16 + lr) * BK, lhi);
#pragma unroll
        for (int j = 0; j < 4; ++j) {
            const v16h b = frag_b(Bb + ((tgl * 4 + j) * 16 + lr) * BK + lhi * 16);
            acc[j] = __builtin_amdgcn_wmma_f32_16x16x32_f16(
                false, a, false, b, (short)0, acc[j], false, false);
        }
    }
#else
    for (int s = 0; s < nsteps; ++s) {
        const int k0 = s * BK;
        // cooperative stage: A = 512 float4 (2/thread), B = 1024 float4 (4/thread)
#pragma unroll
        for (int i = 0; i < 2; ++i) {
            const int v = tid * 2 + i, row = v >> 3, c4 = v & 7;
            *(float4*)&sA[0][row * BK + c4 * 4] =
                *(const float4*)&A[(size_t)(bm0 + row) * lda + k0 + c4 * 4];
        }
#pragma unroll
        for (int i = 0; i < 4; ++i) {
            const int v = tid * 4 + i, row = v >> 3, c4 = v & 7;
            *(float4*)&sB[0][row * BK + c4 * 4] =
                *(const float4*)&W[(size_t)(bn0 + row) * ldw + k0 + c4 * 4];
        }
        __syncthreads();
        const v16h a = frag_a(&sA[0][0] + (tml * 16 + lr) * BK, lhi);
#pragma unroll
        for (int j = 0; j < 4; ++j) {
            const v16h b = frag_b(&sB[0][0] + ((tgl * 4 + j) * 16 + lr) * BK + lhi * 16);
            acc[j] = __builtin_amdgcn_wmma_f32_16x16x32_f16(
                false, a, false, b, (short)0, acc[j], false, false);
        }
        __syncthreads();
    }
#endif

    // store: acc[j][r] -> C[bm0 + tml*16 + r + 8*lhi][bn0 + (tgl*4+j)*16 + lr]
#pragma unroll
    for (int j = 0; j < 4; ++j) {
        const int n = bn0 + (tgl * 4 + j) * 16 + lr;
#pragma unroll
        for (int r = 0; r < 8; ++r) {
            const int m = bm0 + tml * 16 + r + lhi * 8;
            float v = acc[j][r];
            if (EPI == 1) {
                v += bias[n];
                v = (v > 20.0f) ? v : __logf(1.0f + __expf(v));  // softplus
            }
            C[(size_t)m * ldc + n] = v;
        }
    }
}

// ======================================================================
// Direct (non-tiled) WMMA GEMM for small N (x_dbl projection, N=96).
// One wave computes 16 x (16*NT); grouped float4 global loads.
// ======================================================================
template<int NT>
__global__ __launch_bounds__(256)
void wmma_gemm_small(const float* __restrict__ A, int lda,
                     const float* __restrict__ W, int ldw,
                     float* __restrict__ C, int ldc,
                     int M, int N, int K)
{
    const int lane    = threadIdx.x & 31;
    const int wave    = blockIdx.x * 8 + (threadIdx.x >> 5);
    const int tilesM  = M >> 4;
    const int groupsN = (N >> 4) / NT;
    if (wave >= tilesM * groupsN) return;
    const int tm  = wave / groupsN;
    const int tg  = wave % groupsN;
    const int lr  = lane & 15;
    const int lhi = lane >> 4;

    const float* arow = A + (size_t)(tm * 16 + lr) * lda;

    v8f acc[NT];
#pragma unroll
    for (int j = 0; j < NT; ++j) acc[j] = (v8f){0,0,0,0,0,0,0,0};

    for (int k0 = 0; k0 < K; k0 += 32) {
        __builtin_prefetch(arow + k0 + 64, 0, 1);
        const v16h a = frag_a(arow + k0, lhi);
#pragma unroll
        for (int j = 0; j < NT; ++j) {
            const int n0 = (tg * NT + j) * 16;
            const v16h b = frag_b(W + (size_t)(n0 + lr) * ldw + k0 + lhi * 16);
            acc[j] = __builtin_amdgcn_wmma_f32_16x16x32_f16(
                false, a, false, b, (short)0, acc[j], false, false);
        }
    }
#pragma unroll
    for (int j = 0; j < NT; ++j) {
        const int n = (tg * NT + j) * 16 + lr;
#pragma unroll
        for (int r = 0; r < 8; ++r)
            C[(size_t)(tm * 16 + r + lhi * 8) * ldc + n] = acc[j][r];
    }
}

// ======================================================================
// Depthwise causal conv (D_CONV=4) + bias + SiLU  ->  u[b][t][d]
// ======================================================================
__global__ __launch_bounds__(256)
void conv_silu_kernel(const float* __restrict__ xr,
                      const float* __restrict__ cw,
                      const float* __restrict__ cb,
                      float* __restrict__ u)
{
    const long idx = (long)blockIdx.x * blockDim.x + threadIdx.x;
    if (idx >= (long)MROWS * D_INNER) return;
    const int d   = (int)(idx % D_INNER);
    const int row = (int)(idx / D_INNER);
    const int t   = row % SEQ;
    const int b   = row / SEQ;

    float acc = cb[d];
#pragma unroll
    for (int j = 0; j < D_CONV; ++j) {
        const int tt = t - (D_CONV - 1) + j;
        if (tt >= 0)
            acc += xr[((size_t)(b * SEQ + tt)) * (2 * D_INNER) + d] * cw[d * D_CONV + j];
    }
    u[idx] = acc / (1.0f + __expf(-acc));
}

// ======================================================================
// Selective scan: lane owns channel d (16-float state in VGPRs),
// B/C broadcast through LDS, fused D*u + silu(res) gating.
// ======================================================================
__global__ __launch_bounds__(256)
void scan_kernel(const float* __restrict__ xdbl,
                 const float* __restrict__ dt,
                 const float* __restrict__ u,
                 const float* __restrict__ xr,
                 const float* __restrict__ A_log,
                 const float* __restrict__ Dp,
                 float* __restrict__ ymid)
{
    const int d = blockIdx.x * blockDim.x + threadIdx.x;
    const int b = blockIdx.y;

    __shared__ float sBC[2 * D_STATE];

    float An[D_STATE];
#pragma unroll
    for (int n = 0; n < D_STATE; ++n) An[n] = -__expf(A_log[d * D_STATE + n]);
    const float Dd = Dp[d];

    float h[D_STATE];
#pragma unroll
    for (int n = 0; n < D_STATE; ++n) h[n] = 0.0f;

    for (int t = 0; t < SEQ; ++t) {
        const size_t row = (size_t)b * SEQ + t;
        if (threadIdx.x < 2 * D_STATE)
            sBC[threadIdx.x] = xdbl[row * 96 + DT_RANK + threadIdx.x];
        __syncthreads();

        const float dtv  = dt[row * D_INNER + d];
        const float uv   = u [row * D_INNER + d];
        const float resv = xr[row * (2 * D_INNER) + D_INNER + d];

        float y = 0.0f;
#pragma unroll
        for (int n = 0; n < D_STATE; ++n) {
            const float dA = __expf(dtv * An[n]);
            h[n] = dA * h[n] + dtv * sBC[n] * uv;
            y   += h[n] * sBC[D_STATE + n];
        }
        float yv = y + Dd * uv;
        yv *= resv / (1.0f + __expf(-resv));
        ymid[row * D_INNER + d] = yv;
        __syncthreads();
    }
}

// ======================================================================
extern "C" void kernel_launch(void* const* d_in, const int* in_sizes, int n_in,
                              void* d_out, int out_size, void* d_ws, size_t ws_size,
                              hipStream_t stream)
{
    (void)in_sizes; (void)n_in; (void)out_size; (void)ws_size;
    const float* hs     = (const float*)d_in[0];
    const float* W_in   = (const float*)d_in[1];
    const float* conv_w = (const float*)d_in[2];
    const float* conv_b = (const float*)d_in[3];
    const float* W_x    = (const float*)d_in[4];
    const float* W_dt   = (const float*)d_in[5];
    const float* b_dt   = (const float*)d_in[6];
    const float* A_log  = (const float*)d_in[7];
    const float* Dp     = (const float*)d_in[8];
    const float* W_out  = (const float*)d_in[9];
    float* out = (float*)d_out;

    float* ws   = (float*)d_ws;
    float* xr   = ws;                                   // MROWS x 4096
    float* u    = xr   + (size_t)MROWS * 2 * D_INNER;   // MROWS x 2048
    float* xdbl = u    + (size_t)MROWS * D_INNER;       // MROWS x 96
    float* dtb  = xdbl + (size_t)MROWS * 96;            // MROWS x 2048
    float* ymid = dtb  + (size_t)MROWS * D_INNER;       // MROWS x 2048

    // 1) xr = hs @ W_in^T     (4096 x 4096 x 1024)
    wmma_gemm_tiled<0><<<dim3((2 * D_INNER) / 128, MROWS / 64), 256, 0, stream>>>(
        hs, D_MODEL, W_in, D_MODEL, xr, 2 * D_INNER, nullptr,
        MROWS, 2 * D_INNER, D_MODEL);

    // 2) depthwise conv + SiLU -> u
    {
        long total = (long)MROWS * D_INNER;
        conv_silu_kernel<<<(int)((total + 255) / 256), 256, 0, stream>>>(
            xr, conv_w, conv_b, u);
    }

    // 3) x_dbl = u @ W_x^T    (4096 x 96 x 2048), small N -> direct kernel
    {
        long waves = (MROWS / 16) * ((96 / 16) / 2);
        wmma_gemm_small<2><<<(int)((waves + 7) / 8), 256, 0, stream>>>(
            u, D_INNER, W_x, D_INNER, xdbl, 96, MROWS, 96, D_INNER);
    }

    // 4) dt = softplus(x_dbl[:, :64] @ W_dt^T + b_dt)   (4096 x 2048 x 64)
    wmma_gemm_tiled<1><<<dim3(D_INNER / 128, MROWS / 64), 256, 0, stream>>>(
        xdbl, 96, W_dt, DT_RANK, dtb, D_INNER, b_dt,
        MROWS, D_INNER, DT_RANK);

    // 5) selective scan + D*u + silu(res) gating -> ymid
    scan_kernel<<<dim3(D_INNER / 256, BATCH), 256, 0, stream>>>(
        xdbl, dtb, u, xr, A_log, Dp, ymid);

    // 6) out = ymid @ W_out^T  (4096 x 1024 x 2048)
    wmma_gemm_tiled<0><<<dim3(D_MODEL / 128, MROWS / 64), 256, 0, stream>>>(
        ymid, D_INNER, W_out, D_INNER, out, D_MODEL, nullptr,
        MROWS, D_MODEL, D_INNER);
}
